// PairGameHead_31413390803484
// MI455X (gfx1250) — matrix operand order repaired and checked by reading, not compiled, not run on hardware
//
#include <hip/hip_runtime.h>
#include <hip/hip_bf16.h>

typedef __attribute__((ext_vector_type(16))) _Float16 v16h;
typedef __attribute__((ext_vector_type(8)))  float    v8f;

#define NFEAT 64
#define NHID  32
#define NNODE 1024
#define NEDGE 65536
#define THRESH 0.4f

// D = A(16x32 f16) x B(32x16 f16) + C(16x16 f32)
#define WMMA_F16(A, B, C) \
  __builtin_amdgcn_wmma_f32_16x16x32_f16(false, (A), false, (B), (short)0, (C), false, false)

// ---------------------------------------------------------------------------
// Single-instruction DPP reduction step: v += row_shifted(v), invalid lanes -> 0.
// Folds to v_add_f32_dpp (one VALU op, no LDS, no tied-mov).
template <int CTRL>
__device__ __forceinline__ float dpp_add(float v) {
  const int s = __builtin_amdgcn_update_dpp(0, __builtin_bit_cast(int, v),
                                            CTRL, 0xf, 0xf, true);
  return v + __builtin_bit_cast(float, s);
}

// Inclusive scan within each row of 16 lanes; lane 15 / lane 31 hold the row sum.
__device__ __forceinline__ float row16_sum(float v) {
  v = dpp_add<0x111>(v);  // row_shr:1
  v = dpp_add<0x112>(v);  // row_shr:2
  v = dpp_add<0x114>(v);  // row_shr:4
  v = dpp_add<0x118>(v);  // row_shr:8
  return v;
}

// ---------------------------------------------------------------------------
// B fragment: W1 rows [rowbase+K][col], kbase = (lane<16 ? 0 : 8),
// halves h -> K = kbase + (h<8 ? h : h+8), N = col.
__device__ __forceinline__ v16h load_bfrag(const float* __restrict__ W1,
                                           int rowbase, int col, int lane) {
  const int kbase = (lane < 16) ? 0 : 8;
  const float* p = W1 + (size_t)(rowbase + kbase) * NHID + col;
  v16h b;
#pragma unroll
  for (int h = 0; h < 16; ++h) b[h] = (_Float16)p[(h < 8 ? h : h + 8) * NHID];
  return b;
}

struct BFrags { v16h c[2][2]; v16h d[2][2]; };  // [kstep][ntile]

__device__ __forceinline__ void load_all_bfrags(const float* __restrict__ W1,
                                                int m, int lane, BFrags& B) {
#pragma unroll
  for (int ks = 0; ks < 2; ++ks)
#pragma unroll
    for (int nt = 0; nt < 2; ++nt) {
      B.c[ks][nt] = load_bfrag(W1, 2 * NFEAT + ks * 32, nt * 16 + m, lane);  // |xs-xd| block
      B.d[ks][nt] = load_bfrag(W1, 3 * NFEAT + ks * 32, nt * 16 + m, lane);  // xs*xd block
    }
}

// Per-lane slice of an x row (the 32 features this lane feeds into A fragments),
// converted once to packed f16.  dst[fb][h] corresponds to feature
// f = fb*32 + (h<8 ? h : h+8) + koff,  fb = 0,1.
__device__ __forceinline__ void load_xrow_h(const float* __restrict__ xr, int koff,
                                            v16h* dst) {
#pragma unroll
  for (int g = 0; g < 4; ++g) {
    const float4 a = *(const float4*)(xr + koff + g * 16);
    const float4 b = *(const float4*)(xr + koff + g * 16 + 4);
    const int v = g >> 1;
    const int base = (g & 1) * 8;
    dst[v][base + 0] = (_Float16)a.x;
    dst[v][base + 1] = (_Float16)a.y;
    dst[v][base + 2] = (_Float16)a.z;
    dst[v][base + 3] = (_Float16)a.w;
    dst[v][base + 4] = (_Float16)b.x;
    dst[v][base + 5] = (_Float16)b.y;
    dst[v][base + 6] = (_Float16)b.z;
    dst[v][base + 7] = (_Float16)b.w;
  }
}

// A fragments built with packed-f16 math (v_pk_sub/mul + abs mask).
__device__ __forceinline__ void build_frags(const v16h& xi, const v16h& xj,
                                            v16h& fc, v16h& fd) {
  fc = __builtin_elementwise_abs(xi - xj);  // |xs - xd| feature block
  fd = xi * xj;                             // xs * xd  feature block
}

__device__ __forceinline__ float sigmoidf(float v) {
  return 1.0f / (1.0f + __expf(-v));
}

// ---------------------------------------------------------------------------
// Kernel 1: per-node projections.  P1 = x @ W1[0:64] + b1,  P2 = x @ W1[64:128]
__global__ void precompute_proj(const float* __restrict__ x,
                                const float* __restrict__ W1,
                                const float* __restrict__ b1,
                                float* __restrict__ P1,
                                float* __restrict__ P2) {
  const int t = blockIdx.x * blockDim.x + threadIdx.x;
  if (t >= NNODE * NHID) return;
  const int n = t >> 5;
  const int h = t & 31;
  const float* xr = x + (size_t)n * NFEAT;
  float a = b1[h], b = 0.0f;
#pragma unroll 4
  for (int f = 0; f < NFEAT; ++f) {
    const float xv = xr[f];
    a = fmaf(xv, W1[(size_t)f * NHID + h], a);
    b = fmaf(xv, W1[(size_t)(NFEAT + f) * NHID + h], b);
  }
  P1[t] = a;
  P2[t] = b;
}

// ---------------------------------------------------------------------------
// Kernel 2: dense all-pairs branch.  One wave per 16x16 tile of P.
__global__ void __launch_bounds__(256, 1)
dense_pairs(const float* __restrict__ x,
            const float* __restrict__ W1,
            const float* __restrict__ W2,
            const float* __restrict__ b2p,
            const float* __restrict__ P1,
            const float* __restrict__ P2,
            float* __restrict__ Pout) {
  const int lane = threadIdx.x & 31;
  const int wave = threadIdx.x >> 5;
  const int task = blockIdx.x * 8 + wave;      // 4096 tiles
  const int i0 = (task >> 6) * 16;
  const int j0 = (task & 63) * 16;
  const int m    = lane & 15;
  const int koff = (lane < 16) ? 0 : 8;
  const int hi   = (lane < 16) ? 0 : 1;

  BFrags B;
  load_all_bfrags(W1, m, lane, B);

  v16h xi[2];
  load_xrow_h(x + (size_t)(i0 + m) * NFEAT, koff, xi);

  // P1 contribution in C layout (i-tile fixed for whole wave).
  v8f c1lo, c1hi;
#pragma unroll
  for (int r = 0; r < 8; ++r) {
    const float* p1r = P1 + (size_t)(i0 + r + 8 * hi) * NHID;
    c1lo[r] = p1r[m];
    c1hi[r] = p1r[16 + m];
  }
  const float w2lo = W2[m], w2hi = W2[16 + m];
  const float bias2 = b2p[0];

  for (int jj = 0; jj < 16; ++jj) {
    const int j = j0 + jj;
    v16h xj[2];
    load_xrow_h(x + (size_t)j * NFEAT, koff, xj);
    const float p2lo = P2[(size_t)j * NHID + m];
    const float p2hi = P2[(size_t)j * NHID + 16 + m];

    v8f acc0, acc1;
#pragma unroll
    for (int r = 0; r < 8; ++r) { acc0[r] = c1lo[r] + p2lo; acc1[r] = c1hi[r] + p2hi; }

#pragma unroll
    for (int ks = 0; ks < 2; ++ks) {
      v16h fc, fd;
      build_frags(xi[ks], xj[ks], fc, fd);
      acc0 = WMMA_F16(fc, B.c[ks][0], acc0);
      acc1 = WMMA_F16(fc, B.c[ks][1], acc1);
      acc0 = WMMA_F16(fd, B.d[ks][0], acc0);
      acc1 = WMMA_F16(fd, B.d[ks][1], acc1);
    }

    // Layer 2 (ReLU + 32->1) + sigmoid.  Row M = r + 8*hi; DPP scan puts the
    // 16-lane sum in lane 15 (hi=0) / lane 31 (hi=1).
#pragma unroll
    for (int r = 0; r < 8; ++r) {
      float part = fmaxf(acc0[r], 0.0f) * w2lo + fmaxf(acc1[r], 0.0f) * w2hi;
      part = row16_sum(part);
      if (m == 15) {
        const float prob = sigmoidf(part + bias2);
        Pout[(size_t)(i0 + r + 8 * hi) * NNODE + j] = prob;
      }
    }
  }
}

// ---------------------------------------------------------------------------
// Kernel 3: sparse edge branch.  One wave per 16 edges.
__global__ void __launch_bounds__(256, 1)
edge_branch(const float* __restrict__ x,
            const long long* __restrict__ ei,
            const float* __restrict__ W1,
            const float* __restrict__ W2,
            const float* __restrict__ b2p,
            const float* __restrict__ P1,
            const float* __restrict__ P2,
            float* __restrict__ probs,
            float* __restrict__ actions) {
  const int lane = threadIdx.x & 31;
  const int wave = threadIdx.x >> 5;
  const int e0 = (blockIdx.x * 8 + wave) * 16;
  const int m    = lane & 15;
  const int koff = (lane < 16) ? 0 : 8;
  const int hi   = (lane < 16) ? 0 : 1;

  BFrags B;
  load_all_bfrags(W1, m, lane, B);

  const int is = (int)ei[e0 + m];            // src node for A-row m
  const int jd = (int)ei[NEDGE + e0 + m];    // dst node for A-row m
  v16h xi[2], xj[2];
  load_xrow_h(x + (size_t)is * NFEAT, koff, xi);
  load_xrow_h(x + (size_t)jd * NFEAT, koff, xj);

  v8f acc0, acc1;
#pragma unroll
  for (int r = 0; r < 8; ++r) {
    const int e  = e0 + r + 8 * hi;
    const int si = (int)ei[e];
    const int dj = (int)ei[NEDGE + e];
    acc0[r] = P1[(size_t)si * NHID + m]      + P2[(size_t)dj * NHID + m];
    acc1[r] = P1[(size_t)si * NHID + 16 + m] + P2[(size_t)dj * NHID + 16 + m];
  }

#pragma unroll
  for (int ks = 0; ks < 2; ++ks) {
    v16h fc, fd;
    build_frags(xi[ks], xj[ks], fc, fd);
    acc0 = WMMA_F16(fc, B.c[ks][0], acc0);
    acc1 = WMMA_F16(fc, B.c[ks][1], acc1);
    acc0 = WMMA_F16(fd, B.d[ks][0], acc0);
    acc1 = WMMA_F16(fd, B.d[ks][1], acc1);
  }

  const float w2lo = W2[m], w2hi = W2[16 + m];
  const float bias2 = b2p[0];
#pragma unroll
  for (int r = 0; r < 8; ++r) {
    float part = fmaxf(acc0[r], 0.0f) * w2lo + fmaxf(acc1[r], 0.0f) * w2hi;
    part = row16_sum(part);
    if (m == 15) {
      const int e = e0 + r + 8 * hi;
      const float prob = sigmoidf(part + bias2);
      probs[e]   = prob;
      actions[e] = (prob > THRESH) ? 1.0f : 0.0f;
    }
  }
}

// ---------------------------------------------------------------------------
extern "C" void kernel_launch(void* const* d_in, const int* in_sizes, int n_in,
                              void* d_out, int out_size, void* d_ws, size_t ws_size,
                              hipStream_t stream) {
  (void)in_sizes; (void)n_in; (void)out_size; (void)ws_size;
  const float*     x   = (const float*)d_in[0];
  const long long* ei  = (const long long*)d_in[1];   // edge_index (int64)
  /* d_in[2] = edge_index_global: unused by the reference computation */
  const float*     W1  = (const float*)d_in[3];
  const float*     b1  = (const float*)d_in[4];
  const float*     W2  = (const float*)d_in[5];
  const float*     b2  = (const float*)d_in[6];

  float* probs   = (float*)d_out;            // [65536]
  float* actions = probs + NEDGE;            // [65536]
  float* Pout    = actions + NEDGE;          // [1024*1024]

  float* P1 = (float*)d_ws;                  // [1024*32]
  float* P2 = P1 + NNODE * NHID;             // [1024*32]

  precompute_proj<<<(NNODE * NHID + 255) / 256, 256, 0, stream>>>(x, W1, b1, P1, P2);
  dense_pairs<<<512, 256, 0, stream>>>(x, W1, W2, b2, P1, P2, Pout);
  edge_branch<<<512, 256, 0, stream>>>(x, ei, W1, W2, b2, P1, P2, probs, actions);
}
